// AGNNConvLayer_9663676416439
// MI455X (gfx1250) — matrix operand order repaired and checked by reading, not compiled, not run on hardware
//
#include <hip/hip_runtime.h>
#include <hip/hip_bf16.h>
#include <math.h>

typedef __attribute__((ext_vector_type(16))) _Float16 v16h;
typedef __attribute__((ext_vector_type(8)))  _Float16 v8h;
typedef __attribute__((ext_vector_type(8)))  float    v8f;

#define DIM 32
#define EPS_NORM 1e-12f
#define EPS_SM   1e-16f

// ---------------------------------------------------------------------------
// Kernel 1: wave per node, lane = feature dim.
// rnorm[node] = 1/max(||x||,eps); Xh = (x*rnorm) in f16 (WMMA operand feed).
// ---------------------------------------------------------------------------
__global__ void k_normalize(const float* __restrict__ X,
                            _Float16* __restrict__ Xh,
                            float* __restrict__ rnorm,
                            int n_nodes) {
    int node = (int)((blockIdx.x * blockDim.x + threadIdx.x) >> 5);
    int lane = threadIdx.x & 31;
    if (node >= n_nodes) return;
    float x  = X[(size_t)node * DIM + lane];
    float ss = x * x;
    #pragma unroll
    for (int d = 16; d >= 1; d >>= 1) ss += __shfl_xor(ss, d, 32);
    float rn = 1.0f / fmaxf(sqrtf(ss), EPS_NORM);
    Xh[(size_t)node * DIM + lane] = (_Float16)(x * rn);
    if (lane == 0) rnorm[node] = rn;
}

// ---------------------------------------------------------------------------
// Kernel 2: wave per 32 edges = two independent 16-edge WMMA tiles.
// D = Xrow_tile(16x32) x Xcol_tile(32x16); diag(D) = the 16 edge logits.
// Two independent accumulators let the 2nd WMMA + extraction fill the
// 1st WMMA's 4-coexec RAW hazard window (ISA 7.12.1) instead of v_nops,
// and all 8 b128 gathers issue in one clause for L2 latency hiding.
// ---------------------------------------------------------------------------
__device__ __forceinline__ void load_tile_frags(const _Float16* __restrict__ Xh,
                                                const int* __restrict__ row_ids,
                                                const int* __restrict__ col_ids,
                                                long tbase, int lane,
                                                v16h& A, v16h& B) {
    int m = lane & 15;
    int r = row_ids[tbase + m];
    int c = col_ids[tbase + m];
    // A (16x32 f16): lane<16 -> M=lane, K={0..7,16..23}; lane>=16 -> K={8..15,24..31}
    int koffA = (lane < 16) ? 0 : 8;
    // B (32x16 f16): lane<16 -> N=lane, K={0..7,16..23}; lane>=16 -> K={8..15,24..31}
    int koffB = (lane < 16) ? 0 : 8;
    const _Float16* ra = Xh + (size_t)r * DIM + koffA;
    const _Float16* rb = Xh + (size_t)c * DIM + koffB;
    v8h a0 = *(const v8h*)(ra);         // K = koffA .. koffA+7
    v8h a1 = *(const v8h*)(ra + 16);    // K = koffA+16 .. koffA+23
    v8h b0 = *(const v8h*)(rb);         // K = koffB .. koffB+7
    v8h b1 = *(const v8h*)(rb + 16);    // K = koffB+16 .. koffB+23
    #pragma unroll
    for (int i = 0; i < 8; ++i) {
        A[i] = a0[i]; A[i + 8] = a1[i];
        B[i] = b0[i]; B[i + 8] = b1[i];
    }
}

__device__ __forceinline__ float diag_pick(const v8f& C, int lane) {
    // D[i][i]: i<8 -> VGPR i on lane i; i>=8 -> VGPR i-8 on lane i+16.
    int k = lane & 7;
    float ev = C[0];
    #pragma unroll
    for (int i = 1; i < 8; ++i) ev = (k == i) ? C[i] : ev;
    return ev;
}

__global__ void k_edge_logits(const _Float16* __restrict__ Xh,
                              const int* __restrict__ row_ids,
                              const int* __restrict__ col_ids,
                              const float* __restrict__ attw,
                              float* __restrict__ logits,
                              int n_edges) {
    int wave = (int)((blockIdx.x * blockDim.x + threadIdx.x) >> 5);
    int lane = threadIdx.x & 31;
    long base = (long)wave * 32;
    if (base >= n_edges) return;
    float aw = attw[0];

    if (base + 32 <= n_edges) {       // wave-uniform -> EXEC all-ones for WMMA
        v16h A0, B0, A1, B1;
        load_tile_frags(Xh, row_ids, col_ids, base,      lane, A0, B0);
        load_tile_frags(Xh, row_ids, col_ids, base + 16, lane, A1, B1);
        v8f C0 = {}, C1 = {};
        C0 = __builtin_amdgcn_wmma_f32_16x16x32_f16(false, A0, false, B0,
                                                    (short)0, C0, false, false);
        C1 = __builtin_amdgcn_wmma_f32_16x16x32_f16(false, A1, false, B1,
                                                    (short)0, C1, false, false);
        float e0 = diag_pick(C0, lane) * aw;
        float e1 = diag_pick(C1, lane) * aw;
        // lanes 0..7 own edges {lane, 8..15 via lane+24}; store both tiles
        if (lane < 8) {
            logits[base + lane]      = e0;
            logits[base + 16 + lane] = e1;
        } else if (lane >= 24) {
            logits[base + lane - 16] = e0;
            logits[base + lane]      = e1;   // 16 + (lane-16) == lane
        }
    } else {                          // generic tail (not taken when E % 32 == 0)
        long e = base + lane;
        if (e < n_edges) {
            int r = row_ids[e], c = col_ids[e];
            float acc = 0.f;
            #pragma unroll
            for (int kk = 0; kk < DIM; ++kk)
                acc += (float)Xh[(size_t)r * DIM + kk] * (float)Xh[(size_t)c * DIM + kk];
            logits[e] = acc * aw;
        }
    }
}

// ---------------------------------------------------------------------------
// Kernel 3: wave per node. row_ids sorted -> binary search the edge segment.
// Segment max + sum-exp (wave-reduced), then chunked aggregation: 32 edge
// weights computed in parallel, broadcast via shuffles, lane = dim FMA.
// ---------------------------------------------------------------------------
__device__ __forceinline__ int lower_bound_i32(const int* __restrict__ a, int n, int key) {
    int lo = 0, hi = n;
    while (lo < hi) {
        int mid = (lo + hi) >> 1;
        if (a[mid] < key) lo = mid + 1; else hi = mid;
    }
    return lo;
}

__global__ void k_aggregate(const float* __restrict__ X,
                            const float* __restrict__ rnorm,
                            const float* __restrict__ logits,
                            const int* __restrict__ row_ids,
                            const int* __restrict__ col_ids,
                            float* __restrict__ out,
                            int n_nodes, int n_edges) {
    int node = (int)((blockIdx.x * blockDim.x + threadIdx.x) >> 5);
    int lane = threadIdx.x & 31;
    if (node >= n_nodes) return;

    int lo = lower_bound_i32(row_ids, n_edges, node);
    int hi = lower_bound_i32(row_ids, n_edges, node + 1);

    // segment max (lane-strided, coalesced)
    float m = -INFINITY;
    for (int i = lo + lane; i < hi; i += 32) m = fmaxf(m, logits[i]);
    #pragma unroll
    for (int d = 16; d >= 1; d >>= 1) m = fmaxf(m, __shfl_xor(m, d, 32));

    // segment sum of exp
    float s = 0.f;
    for (int i = lo + lane; i < hi; i += 32) s += expf(logits[i] - m);
    #pragma unroll
    for (int d = 16; d >= 1; d >>= 1) s += __shfl_xor(s, d, 32);
    float inv = 1.0f / (s + EPS_SM);

    // aggregation: per 32-edge chunk, lanes compute weights in parallel,
    // then broadcast (shfl) while every lane FMAs its own dim.
    float acc = 0.f;
    for (int e0 = lo; e0 < hi; e0 += 32) {
        int i = e0 + lane;
        float w = 0.f;
        int   c = 0;
        if (i < hi) {
            c = col_ids[i];
            w = expf(logits[i] - m) * rnorm[c];
        }
        int cnt = hi - e0; if (cnt > 32) cnt = 32;   // wave-uniform
        for (int j = 0; j < cnt; ++j) {
            float wj = __shfl(w, j, 32);
            int   cj = __shfl(c, j, 32);
            acc += wj * X[(size_t)cj * DIM + lane];  // 128B coalesced row read
        }
    }
    out[(size_t)node * DIM + lane] = acc * inv;
}

// ---------------------------------------------------------------------------
extern "C" void kernel_launch(void* const* d_in, const int* in_sizes, int n_in,
                              void* d_out, int out_size, void* d_ws, size_t ws_size,
                              hipStream_t stream) {
    const float* X    = (const float*)d_in[0];
    // d_in[1] = weights: unused in the reference's orign=False path
    const float* attw = (const float*)d_in[2];
    const int*   row  = (const int*)d_in[3];
    const int*   col  = (const int*)d_in[4];
    int n_nodes = in_sizes[0] / DIM;
    int n_edges = in_sizes[3];
    float* out  = (float*)d_out;

    // scratch layout: Xh (f16 N*32) | rnorm (f32 N) | logits (f32 E)  (~13.2 MB)
    char* ws = (char*)d_ws;
    _Float16* Xh = (_Float16*)ws;
    size_t off = (size_t)n_nodes * DIM * sizeof(_Float16);
    off = (off + 255) & ~(size_t)255;
    float* rnorm = (float*)(ws + off);
    off += (size_t)n_nodes * sizeof(float);
    off = (off + 255) & ~(size_t)255;
    float* logits = (float*)(ws + off);

    const int BLK = 256;                               // 8 wave32s per block
    int blocksN = (int)(((long)n_nodes * 32 + BLK - 1) / BLK);
    k_normalize<<<blocksN, BLK, 0, stream>>>(X, Xh, rnorm, n_nodes);

    long wavesE = ((long)n_edges + 31) / 32;           // 32 edges (2 WMMA tiles)/wave
    int blocksE = (int)((wavesE * 32 + BLK - 1) / BLK);
    k_edge_logits<<<blocksE, BLK, 0, stream>>>(Xh, row, col, attw, logits, n_edges);

    k_aggregate<<<blocksN, BLK, 0, stream>>>(X, rnorm, logits, row, col, out,
                                             n_nodes, n_edges);
}
